// TaGSim_EdgeLabel_68332929679865
// MI455X (gfx1250) — matrix operand order repaired and checked by reading, not compiled, not run on hardware
//
#include <hip/hip_runtime.h>
#include <hip/hip_bf16.h>
#include <stdint.h>

// ---------- CDNA5 WMMA types ----------
typedef __attribute__((ext_vector_type(16))) __bf16 v16bf;
typedef __attribute__((ext_vector_type(8)))  float  v8f;
typedef __attribute__((ext_vector_type(4)))  float  f4v;

#define NK 6000            // inner GEMM dimension
#define NROWTILES 375      // 6000 / 16
#define NKBLK 188          // ceil(6000/32); last block half-valid
#define KSPLIT0 94         // K-blocks in split 0 (K 0..3007), all full
#define KSPLIT1 94         // K-blocks in split 1 (K 3008..5999 + pad), last is tail
#define PANEL (NKBLK * 32 * 16)   // u16 elements per packed-B panel (one 16-col tile)

__device__ __forceinline__ unsigned int pkbf(float a, float b) {
  // two f32 -> packed bf16x2 (a->[15:0], b->[31:16]), round-to-nearest-ties-away:
  // add 0x8000 then take high halves via a single v_perm_b32.
  const unsigned int ua = __float_as_uint(a) + 0x8000u;
  const unsigned int ub = __float_as_uint(b) + 0x8000u;
  return __builtin_amdgcn_perm(ub, ua, 0x07060302u);
}

union AFrag { unsigned int w[8]; v16bf b; };
union BFrag { uint4 q[2];        v16bf b; };

__device__ __forceinline__ v16bf cvtA(f4v lo0, f4v lo1, f4v hi0, f4v hi1) {
  AFrag af;
  af.w[0] = pkbf(lo0.x, lo0.y); af.w[1] = pkbf(lo0.z, lo0.w);
  af.w[2] = pkbf(lo1.x, lo1.y); af.w[3] = pkbf(lo1.z, lo1.w);
  af.w[4] = pkbf(hi0.x, hi0.y); af.w[5] = pkbf(hi0.z, hi0.w);
  af.w[6] = pkbf(hi1.x, hi1.y); af.w[7] = pkbf(hi1.z, hi1.w);
  return af.b;
}

// ================= B pre-pack =================
// Per-lane WMMA B-fragment stream for one 16-column tile, zero-filled past K=5999.
// Optional second source is summed in (deterministic reduction of K-split partials).
struct PackArgs {
  const float*    src[6];
  const float*    src2[6];   // nullptr if none
  unsigned short* dst[6];
  int             ncols[6];
  int             ntile[6];
};

__global__ __launch_bounds__(256) void packB_kernel(PackArgs pa) {
  const int panel = blockIdx.x / 24;
  const int chunk = blockIdx.x % 24;
  const int t = chunk * 256 + threadIdx.x;       // (kblk, lane) index
  if (t >= NKBLK * 32) return;
  const int kblk = t >> 5;
  const int lane = t & 31;
  const int half = lane >> 4;
  const int lm   = lane & 15;

  const float* __restrict__ src  = pa.src[panel];
  const float* __restrict__ src2 = pa.src2[panel];
  const int ncols = pa.ncols[panel];
  const int n = pa.ntile[panel] * 16 + lm;
  const int kb = kblk * 32 + half * 8;

  float v[16];
#pragma unroll
  for (int j = 0; j < 16; ++j) {
    const int k = kb + (j < 8 ? j : j + 8);
    float x = 0.f;
    if (k < NK) {
      x = src[(size_t)k * ncols + n];
      if (src2) x += src2[(size_t)k * ncols + n];
    }
    v[j] = x;
  }
  unsigned int w[8];
#pragma unroll
  for (int j = 0; j < 8; ++j) w[j] = pkbf(v[2 * j], v[2 * j + 1]);

  unsigned short* d = pa.dst[panel] + (size_t)(kblk * 32 + lane) * 16;
  *(uint4*)(d)     = make_uint4(w[0], w[1], w[2], w[3]);
  *(uint4*)(d + 8) = make_uint4(w[4], w[5], w[6], w[7]);
}

// ================= GEMM =================
// One wave per block: one 16x16 output tile, half of K (deterministic 2-way
// K-split into partial buffers). Software-pipelined: loads for k+1 issued
// before the convert+WMMA of k. A streamed with non-temporal loads.
struct GemmArgs {
  const float*          adj[4];
  const unsigned short* pb[4];   // packed-B base; panel nTile at +nTile*PANEL
  float*                out0[4]; // K-split 0 partial
  float*                out1[4]; // K-split 1 partial
  int                   ncols[4];
  int                   nblocks[4];  // 375 * nct * 2
  int                   ntasks;
};

__global__ __launch_bounds__(32) void gemm_bf16_wmma(GemmArgs ga) {
  int bid = blockIdx.x;
  int task = 0;
  while (task < ga.ntasks - 1 && bid >= ga.nblocks[task]) { bid -= ga.nblocks[task]; ++task; }

  const float* __restrict__ adj = ga.adj[task];
  const int ncols = ga.ncols[task];
  const int nct   = ncols >> 4;
  const int per   = NROWTILES * nct;

  const int ksplit  = bid / per;
  const int rem     = bid % per;
  const int nTile   = rem / NROWTILES;
  const int rowTile = rem % NROWTILES;
  float* __restrict__ out = ksplit ? ga.out1[task] : ga.out0[task];

  const int lane = threadIdx.x;    // 0..31 (wave32)
  const int half = lane >> 4;
  const int lm   = lane & 15;
  const int koffA = half * 8;      // A-matrix K sub-group per 16-bit A layout

  const int kb0   = ksplit ? KSPLIT0 : 0;
  const int nfull = ksplit ? (KSPLIT1 - 1) : KSPLIT0;  // full K-blocks in this split

  const float* ap = adj + (size_t)(rowTile * 16 + lm) * NK + koffA + (size_t)kb0 * 32;
  const unsigned short* blp =
      ga.pb[task] + (size_t)nTile * PANEL + (size_t)lane * 16 + (size_t)kb0 * 512;

  v8f acc = {};

  // ---- prologue: load K-block 0 of this split ----
  uint4 b0 = *(const uint4*)(blp);
  uint4 b1 = *(const uint4*)(blp + 8);
  f4v a0 = __builtin_nontemporal_load((const f4v*)(ap));
  f4v a1 = __builtin_nontemporal_load((const f4v*)(ap + 4));
  f4v a2 = __builtin_nontemporal_load((const f4v*)(ap + 16));
  f4v a3 = __builtin_nontemporal_load((const f4v*)(ap + 20));

#pragma unroll 1
  for (int i = 1; i < nfull; ++i) {
    // loads for next K-block
    const float*          apn = ap  + (size_t)i * 32;
    const unsigned short* bln = blp + (size_t)i * 512;
    const uint4 nb0 = *(const uint4*)(bln);
    const uint4 nb1 = *(const uint4*)(bln + 8);
    const f4v n0 = __builtin_nontemporal_load((const f4v*)(apn));
    const f4v n1 = __builtin_nontemporal_load((const f4v*)(apn + 4));
    const f4v n2 = __builtin_nontemporal_load((const f4v*)(apn + 16));
    const f4v n3 = __builtin_nontemporal_load((const f4v*)(apn + 20));

    // compute current K-block
    BFrag bf; bf.q[0] = b0; bf.q[1] = b1;
    acc = __builtin_amdgcn_wmma_f32_16x16x32_bf16(false, cvtA(a0, a1, a2, a3),
                                                  false, bf.b, (short)0, acc, false, false);
    b0 = nb0; b1 = nb1; a0 = n0; a1 = n1; a2 = n2; a3 = n3;
  }

  if (ksplit) {
    // tail K-block 187 (K 5984..5999 valid): load low halves + packed B (zero-padded)
    const float*          apt = ap  + (size_t)nfull * 32;
    const unsigned short* blt = blp + (size_t)nfull * 512;
    const uint4 tb0 = *(const uint4*)(blt);
    const uint4 tb1 = *(const uint4*)(blt + 8);
    const f4v t0 = __builtin_nontemporal_load((const f4v*)(apt));
    const f4v t1 = __builtin_nontemporal_load((const f4v*)(apt + 4));

    // last full block
    BFrag bf; bf.q[0] = b0; bf.q[1] = b1;
    acc = __builtin_amdgcn_wmma_f32_16x16x32_bf16(false, cvtA(a0, a1, a2, a3),
                                                  false, bf.b, (short)0, acc, false, false);
    // tail block
    AFrag af;
    af.w[0] = pkbf(t0.x, t0.y); af.w[1] = pkbf(t0.z, t0.w);
    af.w[2] = pkbf(t1.x, t1.y); af.w[3] = pkbf(t1.z, t1.w);
    af.w[4] = 0u; af.w[5] = 0u; af.w[6] = 0u; af.w[7] = 0u;
    BFrag bft; bft.q[0] = tb0; bft.q[1] = tb1;
    acc = __builtin_amdgcn_wmma_f32_16x16x32_bf16(false, af.b,
                                                  false, bft.b, (short)0, acc, false, false);
  } else {
    BFrag bf; bf.q[0] = b0; bf.q[1] = b1;
    acc = __builtin_amdgcn_wmma_f32_16x16x32_bf16(false, cvtA(a0, a1, a2, a3),
                                                  false, bf.b, (short)0, acc, false, false);
  }

  // ---- store D: VGPR r holds M = r + 8*half, N = lm ----
  const int colBase = nTile * 16 + lm;
  const int row0 = rowTile * 16 + half * 8;
#pragma unroll
  for (int r = 0; r < 8; ++r)
    out[(size_t)(row0 + r) * ncols + colBase] = acc[r];
}

// ---------------- column-sum pooling ----------------
struct CsTask { const float* src; const float* src2; int cols; int dst; };
struct CsArgs { CsTask t[10]; };

__global__ __launch_bounds__(256) void zero_cs_kernel(float* cs) {
  cs[threadIdx.x] = 0.f;   // 256 pooled floats
}

__global__ __launch_bounds__(256) void colsum_kernel(CsArgs ca, float* __restrict__ cs) {
  const int mat   = blockIdx.x / 10;
  const int chunk = blockIdx.x % 10;      // 600 rows per chunk
  const CsTask T = ca.t[mat];
  const int cols = T.cols;                // 16 or 32 (divides 256)
  const int tid  = threadIdx.x;
  const int col  = tid % cols;
  const int g    = tid / cols;
  const int groups = 256 / cols;
  const int rEnd = chunk * 600 + 600;

  float sum = 0.f;
  for (int r = chunk * 600 + g; r < rEnd; r += groups) {
    float x = T.src[(size_t)r * cols + col];
    if (T.src2) x += T.src2[(size_t)r * cols + col];
    sum += x;
  }

  __shared__ float sm[256];
  sm[tid] = sum;
  __syncthreads();
  if (tid < cols) {
    float tot = 0.f;
    for (int gg = 0; gg < groups; ++gg) tot += sm[gg * cols + tid];
    atomicAdd(cs + T.dst + tid, tot);
  }
}

// ---------------- head: tensor networks + MLPs + score ----------------
__global__ __launch_bounds__(64) void head_kernel(
    const float* __restrict__ cs, const float* __restrict__ avg_v,
    const float* __restrict__ Wn, const float* __restrict__ Wbn, const float* __restrict__ bn,
    const float* __restrict__ We, const float* __restrict__ Wbe, const float* __restrict__ be,
    const float* __restrict__ m1w, const float* __restrict__ m1b,
    const float* __restrict__ m2w, const float* __restrict__ m2b,
    const float* __restrict__ m3w, const float* __restrict__ m3b,
    const float* __restrict__ sw,  const float* __restrict__ sb,
    float* __restrict__ out) {
  __shared__ float e1v[4][64], e2v[4][64];
  __shared__ float s[4][16], h1[4][16], h2s[4][8], h3[4][4], sc[4];
  const int tid = threadIdx.x;

  // pooled concat vectors (cs layout: emb1@0, emb2@32, g1h1@64, g1h2@96,
  // g2h1@128, g2h2@160, ea1@192, ea2@208, e1h1@224, e2h1@240)
  {
    const int d = tid;
    const float g1_01 = (d < 32) ? cs[0 + d]   : cs[64  + (d - 32)];
    const float g2_01 = (d < 32) ? cs[32 + d]  : cs[128 + (d - 32)];
    const float g1_12 = (d < 32) ? cs[64 + d]  : cs[96  + (d - 32)];
    const float g2_12 = (d < 32) ? cs[128 + d] : cs[160 + (d - 32)];
    e1v[0][d] = g1_01; e2v[0][d] = g2_01;
    e1v[1][d] = g1_01; e2v[1][d] = g2_01;
    e1v[2][d] = g1_12; e2v[2][d] = g2_12;
    if (d < 32) {
      e1v[3][d] = (d < 16) ? cs[192 + d] : cs[224 + (d - 16)];
      e2v[3][d] = (d < 16) ? cs[208 + d] : cs[240 + (d - 16)];
    }
  }
  __syncthreads();

  // s[b][k] = relu( e1 . W[:,:,k] . e2 + Wb[k] @ [e1;e2] + bias )
  {
    const int b = tid >> 4, k = tid & 15;
    float acc;
    if (b < 3) {
      const float* W = Wn + (size_t)b * 64 * 64 * 16;
      float sco = 0.f;
      for (int d = 0; d < 64; ++d) {
        const float e1d = e1v[b][d];
        const float* Wd = W + (size_t)d * 64 * 16 + k;
        for (int e = 0; e < 64; ++e) sco += e1d * Wd[(size_t)e * 16] * e2v[b][e];
      }
      float blk = 0.f;
      const float* Wb = Wbn + (size_t)(b * 16 + k) * 128;
      for (int j = 0; j < 64; ++j) blk += Wb[j]      * e1v[b][j];
      for (int j = 0; j < 64; ++j) blk += Wb[64 + j] * e2v[b][j];
      acc = sco + blk + bn[b * 16 + k];
    } else {
      float sco = 0.f;
      for (int d = 0; d < 32; ++d) {
        const float e1d = e1v[3][d];
        const float* Wd = We + (size_t)d * 32 * 16 + k;
        for (int e = 0; e < 32; ++e) sco += e1d * Wd[(size_t)e * 16] * e2v[3][e];
      }
      float blk = 0.f;
      const float* Wb = Wbe + (size_t)k * 64;
      for (int j = 0; j < 32; ++j) blk += Wb[j]      * e1v[3][j];
      for (int j = 0; j < 32; ++j) blk += Wb[32 + j] * e2v[3][j];
      acc = sco + blk + be[k];
    }
    s[b][k] = fmaxf(acc, 0.f);
  }
  __syncthreads();

  { // MLP1: 16 -> 16
    const int b = tid >> 4, o = tid & 15;
    float a = m1b[b * 16 + o];
    for (int i = 0; i < 16; ++i) a += m1w[(b * 16 + o) * 16 + i] * s[b][i];
    h1[b][o] = fmaxf(a, 0.f);
  }
  __syncthreads();
  if (tid < 32) { // MLP2: 16 -> 8
    const int b = tid >> 3, o = tid & 7;
    float a = m2b[b * 8 + o];
    for (int i = 0; i < 16; ++i) a += m2w[(b * 8 + o) * 16 + i] * h1[b][i];
    h2s[b][o] = fmaxf(a, 0.f);
  }
  __syncthreads();
  if (tid < 16) { // MLP3: 8 -> 4
    const int b = tid >> 2, o = tid & 3;
    float a = m3b[b * 4 + o];
    for (int i = 0; i < 8; ++i) a += m3w[(b * 4 + o) * 8 + i] * h2s[b][i];
    h3[b][o] = fmaxf(a, 0.f);
  }
  __syncthreads();
  if (tid < 4) { // score head
    float a = sb[tid];
    for (int i = 0; i < 4; ++i) a += sw[tid * 4 + i] * h3[tid][i];
    sc[tid] = 1.f / (1.f + __expf(-a));
  }
  __syncthreads();
  if (tid == 0) {
    const float so[4] = { sc[0], sc[1], sc[3], sc[2] };  // order: nc, in, ec, ie
    const float av = avg_v[0];
    float ged = 0.f;
    for (int i = 0; i < 4; ++i) ged += -__logf(so[i]);
    ged *= av;
    out[0] = __expf(-ged / av);  // pre_sim
    out[1] = ged;                // pre_ged
    out[2] = so[0]; out[3] = so[1]; out[4] = so[2]; out[5] = so[3];
  }
}

extern "C" void kernel_launch(void* const* d_in, const int* in_sizes, int n_in,
                              void* d_out, int out_size, void* d_ws, size_t ws_size,
                              hipStream_t stream) {
  const float* emb1       = (const float*)d_in[0];
  const float* emb2       = (const float*)d_in[1];
  const float* adj1       = (const float*)d_in[2];
  const float* adj2       = (const float*)d_in[3];
  const float* edge_attr1 = (const float*)d_in[4];
  const float* edge_attr2 = (const float*)d_in[5];
  const float* edge_adj1  = (const float*)d_in[6];
  const float* edge_adj2  = (const float*)d_in[7];
  const float* avg_v      = (const float*)d_in[8];
  const float* Wn         = (const float*)d_in[9];
  const float* Wbn        = (const float*)d_in[10];
  const float* bn         = (const float*)d_in[11];
  const float* We         = (const float*)d_in[12];
  const float* Wbe        = (const float*)d_in[13];
  const float* be         = (const float*)d_in[14];
  const float* m1w        = (const float*)d_in[15];
  const float* m1b        = (const float*)d_in[16];
  const float* m2w        = (const float*)d_in[17];
  const float* m2b        = (const float*)d_in[18];
  const float* m3w        = (const float*)d_in[19];
  const float* m3b        = (const float*)d_in[20];
  const float* sw         = (const float*)d_in[21];
  const float* sb         = (const float*)d_in[22];
  float* out = (float*)d_out;

  // workspace layout (floats); two K-split partial buffers per product
  float* ws      = (float*)d_ws;
  float* g1h1_p0 = ws + 0;        float* g1h1_p1 = ws + 192000;
  float* g2h1_p0 = ws + 384000;   float* g2h1_p1 = ws + 576000;
  float* g1h2_p0 = ws + 768000;   float* g1h2_p1 = ws + 960000;
  float* g2h2_p0 = ws + 1152000;  float* g2h2_p1 = ws + 1344000;
  float* e1h1_p0 = ws + 1536000;  float* e1h1_p1 = ws + 1632000;
  float* e2h1_p0 = ws + 1728000;  float* e2h1_p1 = ws + 1824000;
  float* cs      = ws + 1920000;  // 256 pooled floats
  unsigned short* pbase = (unsigned short*)(ws + 1920256);
  unsigned short* pb_emb1 = pbase;                  // 2 panels
  unsigned short* pb_emb2 = pbase + 2 * PANEL;      // 2 panels
  unsigned short* pb_ea1  = pbase + 4 * PANEL;      // 1 panel
  unsigned short* pb_ea2  = pbase + 5 * PANEL;      // 1 panel
  unsigned short* pb_g1h1 = pbase + 6 * PANEL;      // 2 panels
  unsigned short* pb_g2h1 = pbase + 8 * PANEL;      // 2 panels

  // ---- pre-pack B panels for pass 1 ----
  PackArgs p1{};
  p1.src[0] = emb1;       p1.dst[0] = pb_emb1;         p1.ncols[0] = 32; p1.ntile[0] = 0;
  p1.src[1] = emb1;       p1.dst[1] = pb_emb1 + PANEL; p1.ncols[1] = 32; p1.ntile[1] = 1;
  p1.src[2] = emb2;       p1.dst[2] = pb_emb2;         p1.ncols[2] = 32; p1.ntile[2] = 0;
  p1.src[3] = emb2;       p1.dst[3] = pb_emb2 + PANEL; p1.ncols[3] = 32; p1.ntile[3] = 1;
  p1.src[4] = edge_attr1; p1.dst[4] = pb_ea1;          p1.ncols[4] = 16; p1.ntile[4] = 0;
  p1.src[5] = edge_attr2; p1.dst[5] = pb_ea2;          p1.ncols[5] = 16; p1.ntile[5] = 0;
  packB_kernel<<<6 * 24, 256, 0, stream>>>(p1);

  // ---- pass 1: h1 partials = adj @ feat ----
  GemmArgs g1{};
  g1.adj[0] = adj1;      g1.pb[0] = pb_emb1; g1.out0[0] = g1h1_p0; g1.out1[0] = g1h1_p1; g1.ncols[0] = 32; g1.nblocks[0] = NROWTILES * 4;
  g1.adj[1] = adj2;      g1.pb[1] = pb_emb2; g1.out0[1] = g2h1_p0; g1.out1[1] = g2h1_p1; g1.ncols[1] = 32; g1.nblocks[1] = NROWTILES * 4;
  g1.adj[2] = edge_adj1; g1.pb[2] = pb_ea1;  g1.out0[2] = e1h1_p0; g1.out1[2] = e1h1_p1; g1.ncols[2] = 16; g1.nblocks[2] = NROWTILES * 2;
  g1.adj[3] = edge_adj2; g1.pb[3] = pb_ea2;  g1.out0[3] = e2h1_p0; g1.out1[3] = e2h1_p1; g1.ncols[3] = 16; g1.nblocks[3] = NROWTILES * 2;
  g1.ntasks = 4;
  gemm_bf16_wmma<<<NROWTILES * 12, 32, 0, stream>>>(g1);

  // ---- pack h1 (= p0 + p1, deterministic) as B panels for pass 2 ----
  PackArgs p2{};
  p2.src[0] = g1h1_p0; p2.src2[0] = g1h1_p1; p2.dst[0] = pb_g1h1;         p2.ncols[0] = 32; p2.ntile[0] = 0;
  p2.src[1] = g1h1_p0; p2.src2[1] = g1h1_p1; p2.dst[1] = pb_g1h1 + PANEL; p2.ncols[1] = 32; p2.ntile[1] = 1;
  p2.src[2] = g2h1_p0; p2.src2[2] = g2h1_p1; p2.dst[2] = pb_g2h1;         p2.ncols[2] = 32; p2.ntile[2] = 0;
  p2.src[3] = g2h1_p0; p2.src2[3] = g2h1_p1; p2.dst[3] = pb_g2h1 + PANEL; p2.ncols[3] = 32; p2.ntile[3] = 1;
  packB_kernel<<<4 * 24, 256, 0, stream>>>(p2);

  // ---- pass 2: h2 partials = adj @ h1 (node graphs only) ----
  GemmArgs g2{};
  g2.adj[0] = adj1; g2.pb[0] = pb_g1h1; g2.out0[0] = g1h2_p0; g2.out1[0] = g1h2_p1; g2.ncols[0] = 32; g2.nblocks[0] = NROWTILES * 4;
  g2.adj[1] = adj2; g2.pb[1] = pb_g2h1; g2.out0[1] = g2h2_p0; g2.out1[1] = g2h2_p1; g2.ncols[1] = 32; g2.nblocks[1] = NROWTILES * 4;
  g2.ntasks = 2;
  gemm_bf16_wmma<<<NROWTILES * 8, 32, 0, stream>>>(g2);

  // ---- pooling: column sums (K-split partials summed in-kernel) ----
  zero_cs_kernel<<<1, 256, 0, stream>>>(cs);
  CsArgs ca{};
  ca.t[0] = {emb1,       nullptr, 32, 0};
  ca.t[1] = {emb2,       nullptr, 32, 32};
  ca.t[2] = {g1h1_p0,    g1h1_p1, 32, 64};
  ca.t[3] = {g1h2_p0,    g1h2_p1, 32, 96};
  ca.t[4] = {g2h1_p0,    g2h1_p1, 32, 128};
  ca.t[5] = {g2h2_p0,    g2h2_p1, 32, 160};
  ca.t[6] = {edge_attr1, nullptr, 16, 192};
  ca.t[7] = {edge_attr2, nullptr, 16, 208};
  ca.t[8] = {e1h1_p0,    e1h1_p1, 16, 224};
  ca.t[9] = {e2h1_p0,    e2h1_p1, 16, 240};
  colsum_kernel<<<100, 256, 0, stream>>>(ca, cs);

  // ---- head ----
  head_kernel<<<1, 64, 0, stream>>>(cs, avg_v, Wn, Wbn, bn, We, Wbe, be,
                                    m1w, m1b, m2w, m2b, m3w, m3b, sw, sb, out);

  (void)in_sizes; (void)n_in; (void)out_size; (void)ws_size;
}
